// CausalSelfAttention_67800353735026
// MI455X (gfx1250) — matrix operand order repaired and checked by reading, not compiled, not run on hardware
//
#include <hip/hip_runtime.h>
#include <hip/hip_bf16.h>

#define B_  2
#define T_  2048
#define D_  2048
#define H_  16
#define HD_ 128

typedef __bf16 v16bf __attribute__((ext_vector_type(16)));
typedef float  v8f   __attribute__((ext_vector_type(8)));

union frag16 {
    v16bf v;
    uint4 q[2];
    unsigned short u[16];
};

static __device__ __forceinline__ unsigned short f2bf(float f) {
    union { float f; unsigned u; } c; c.f = f;
    unsigned u = c.u;
    u += 0x7FFFu + ((u >> 16) & 1u);   // round-to-nearest-even
    return (unsigned short)(u >> 16);
}

static __device__ __forceinline__ v8f wmma_bf16(const frag16& a, const frag16& b, v8f c) {
    return __builtin_amdgcn_wmma_f32_16x16x32_bf16(
        /*neg_a=*/false, a.v, /*neg_b=*/false, b.v,
        /*c_mod=*/(short)0, c, /*reuse_a=*/false, /*reuse_b=*/false);
}

// A-fragment (16x32 bf16, row-major source, lda in elements).
// ISA 16-bit A layout: lanes hold row m=lane%16; VGPR0-3: K = 8*half+0..7,
// VGPR4-7: K = 16+8*half+0..7  -> two contiguous 16B loads per lane.
static __device__ __forceinline__ frag16 load_a_frag(const unsigned short* base,
                                                     int lda, int k0, int lane) {
    int m = lane & 15, h = lane >> 4;
    const unsigned short* p = base + m * lda + k0 + h * 8;
    frag16 r;
    r.q[0] = *(const uint4*)(p);
    r.q[1] = *(const uint4*)(p + 16);
    return r;
}

// B-fragment (32x16 bf16) from a K-contiguous layout: src[n][k], row stride ldk.
// ISA B layout: lane = column n; elements 0-7: K = k0+8*half+0..7,
// elements 8-15: K = k0+16+8*half+0..7 -> two contiguous 16B reads per lane.
static __device__ __forceinline__ frag16 load_b_frag(const unsigned short* base,
                                                     int ldk, int k0, int lane) {
    int n = lane & 15, h = lane >> 4;
    const unsigned short* p = base + n * ldk + k0 + h * 8;
    frag16 r;
    r.q[0] = *(const uint4*)(p);
    r.q[1] = *(const uint4*)(p + 16);
    return r;
}

// ---------------- conversion kernels ----------------
__global__ void cvt_bf16(const float* __restrict__ x, unsigned short* __restrict__ y, int n) {
    for (int i = blockIdx.x * blockDim.x + threadIdx.x; i < n; i += gridDim.x * blockDim.x)
        y[i] = f2bf(x[i]);
}

// w is [K,N] fp32; write wt as [N,K] bf16 so B-fragments are contiguous.
__global__ void cvt_transpose(const float* __restrict__ w, unsigned short* __restrict__ wt,
                              int K, int N) {
    int total = K * N;
    for (int i = blockIdx.x * blockDim.x + threadIdx.x; i < total; i += gridDim.x * blockDim.x) {
        int k = i / N, n = i - k * N;
        wt[n * K + k] = f2bf(w[i]);
    }
}

// ---------------- QKV GEMM + fused RoPE ----------------
// One wave computes a 16-row x 128-col tile (one head-slice of q, k or v).
// tasks = 256 m-tiles * 48 head-slices = 12288; 4 waves per block.
__global__ __launch_bounds__(128) void gemm_qkv_rope(
    const unsigned short* __restrict__ xbf,   // [B*T, D] bf16
    const unsigned short* __restrict__ wT,    // [3D, D] bf16 (transposed)
    unsigned short* __restrict__ Qb,          // [B,H,T,Hd]
    unsigned short* __restrict__ Kb,          // [B,H,T,Hd]
    unsigned short* __restrict__ Vt)          // [B,H,Hd,T]
{
    int lane = threadIdx.x & 31, wid = threadIdx.x >> 5;
    int task = blockIdx.x * 4 + wid;
    int mt = task & 255;          // 256 m-tiles of 16 rows
    int hs = task >> 8;           // 48 head-slices (16 q, 16 k, 16 v)
    int m0 = mt * 16;
    int sec = hs >> 4, h = hs & 15;
    int nbase = hs * 128;

    v8f zero = {0.f,0.f,0.f,0.f,0.f,0.f,0.f,0.f};
    v8f acc[8];
#pragma unroll
    for (int j = 0; j < 8; ++j) acc[j] = zero;

    const unsigned short* abase = xbf + m0 * D_;
    for (int kc = 0; kc < D_ / 32; ++kc) {
        if (kc + 1 < D_ / 32)
            __builtin_prefetch(abase + (lane & 15) * D_ + (kc + 1) * 32, 0, 1);
        frag16 a = load_a_frag(abase, D_, kc * 32, lane);
#pragma unroll
        for (int j = 0; j < 8; ++j) {
            frag16 b = load_b_frag(wT + (nbase + j * 16) * D_, D_, kc * 32, lane);
            acc[j] = wmma_bf16(a, b, acc[j]);
        }
    }

    int half = lane >> 4, nc = lane & 15;
    int bidx = m0 >> 11;          // batch (T_ == 2048)
    int tb = m0 & (T_ - 1);

    if (sec == 2) {
        // V: store transposed [Hd, T]
        unsigned short* vdst = Vt + (size_t)((bidx * H_ + h) * HD_) * T_;
#pragma unroll
        for (int j = 0; j < 8; ++j) {
            int d = j * 16 + nc;
#pragma unroll
            for (int r = 0; r < 8; ++r) {
                int t = tb + r + 8 * half;
                vdst[d * T_ + t] = f2bf(acc[j][r]);
            }
        }
    } else {
        // Q/K: fused RoPE. Tiles j (d<64) and j+4 (d+64) are rotate-half partners
        // and share the same (lane, vgpr-row) slot in the C layout.
        unsigned short* dst = (sec == 0 ? Qb : Kb) + (size_t)((bidx * H_ + h) * T_) * HD_;
#pragma unroll
        for (int j = 0; j < 4; ++j) {
            int d = j * 16 + nc;                               // 0..63
            float inv_freq = __powf(10000.0f, -(float)d * (1.0f / 64.0f));
#pragma unroll
            for (int r = 0; r < 8; ++r) {
                int t = tb + r + 8 * half;
                float sv, cv;
                __sincosf((float)t * inv_freq, &sv, &cv);
                float x1 = acc[j][r];        // column d
                float x2 = acc[j + 4][r];    // column d+64
                unsigned short* row = dst + t * HD_;
                row[d]      = f2bf(x1 * cv - x2 * sv);
                row[d + 64] = f2bf(x2 * cv + x1 * sv);
            }
        }
    }
}

// ---------------- flash attention ----------------
// One wave per (b, h, 16-query tile). tasks = B*H*(T/16) = 4096; 4 waves/block.
__global__ __launch_bounds__(128) void attn_fwd(
    const unsigned short* __restrict__ Qb,    // [B,H,T,Hd]
    const unsigned short* __restrict__ Kb,    // [B,H,T,Hd]
    const unsigned short* __restrict__ Vt,    // [B,H,Hd,T]
    unsigned short* __restrict__ Ob)          // [B,T,D] bf16 (heads concatenated)
{
    __shared__ unsigned short pbuf[4][16 * 64];   // per-wave 16x32 P staging (stride 64)
    int lane = threadIdx.x & 31, wid = threadIdx.x >> 5;
    int task = blockIdx.x * 4 + wid;
    int bh = task >> 7;          // 0..31  (B*H)
    int qt = task & 127;         // 128 q-tiles
    int q0 = qt * 16;
    int half = lane >> 4, nc = lane & 15;

    const unsigned short* qbase = Qb + (size_t)(bh * T_ + q0) * HD_;
    frag16 qf[4];
#pragma unroll
    for (int kc = 0; kc < 4; ++kc) qf[kc] = load_a_frag(qbase, HD_, kc * 32, lane);

    v8f zero = {0.f,0.f,0.f,0.f,0.f,0.f,0.f,0.f};
    v8f o[8];
    float mr[8], lr[8];
#pragma unroll
    for (int j = 0; j < 8; ++j) o[j] = zero;
#pragma unroll
    for (int r = 0; r < 8; ++r) { mr[r] = -3.0e38f; lr[r] = 0.f; }

    const float scale = 0.08838834764831845f;   // 1/sqrt(128)
    int nkb = (q0 + 16 + 31) >> 5;

    for (int kb = 0; kb < nkb; ++kb) {
        const unsigned short* kp = Kb + (size_t)(bh * T_ + kb * 32) * HD_;
        v8f s0 = zero, s1 = zero;
#pragma unroll
        for (int kc = 0; kc < 4; ++kc) {
            frag16 b0 = load_b_frag(kp, HD_, kc * 32, lane);
            s0 = wmma_bf16(qf[kc], b0, s0);
            frag16 b1 = load_b_frag(kp + 16 * HD_, HD_, kc * 32, lane);
            s1 = wmma_bf16(qf[kc], b1, s1);
        }
        int key0 = kb * 32 + nc, key1 = key0 + 16;
#pragma unroll
        for (int r = 0; r < 8; ++r) {
            int q = q0 + r + 8 * half;
            float v0 = (key0 <= q) ? s0[r] * scale : -1.0e30f;
            float v1 = (key1 <= q) ? s1[r] * scale : -1.0e30f;
            float mx = fmaxf(v0, v1);
            mx = fmaxf(mx, __shfl_xor(mx, 1, 16));
            mx = fmaxf(mx, __shfl_xor(mx, 2, 16));
            mx = fmaxf(mx, __shfl_xor(mx, 4, 16));
            mx = fmaxf(mx, __shfl_xor(mx, 8, 16));
            float mnew  = fmaxf(mr[r], mx);
            float alpha = __expf(mr[r] - mnew);
            float p0 = __expf(v0 - mnew);
            float p1 = __expf(v1 - mnew);
            float rs = p0 + p1;
            rs += __shfl_xor(rs, 1, 16);
            rs += __shfl_xor(rs, 2, 16);
            rs += __shfl_xor(rs, 4, 16);
            rs += __shfl_xor(rs, 8, 16);
            lr[r] = lr[r] * alpha + rs;
            mr[r] = mnew;
#pragma unroll
            for (int j = 0; j < 8; ++j) o[j][r] *= alpha;
            int row = r + 8 * half;
            pbuf[wid][row * 64 + nc]      = f2bf(p0);   // C-layout -> LDS
            pbuf[wid][row * 64 + 16 + nc] = f2bf(p1);
        }
        // cross-lane LDS RAW inside one wave: drain DS before re-reading transposed
        asm volatile("s_wait_dscnt 0x0" ::: "memory");
        frag16 pf;
        {
            const unsigned short* pp = &pbuf[wid][nc * 64 + half * 8];
            pf.q[0] = *(const uint4*)(pp);
            pf.q[1] = *(const uint4*)(pp + 16);
        }
        const unsigned short* vb = Vt + (size_t)(bh * HD_) * T_ + kb * 32;
#pragma unroll
        for (int j = 0; j < 8; ++j) {
            frag16 bv;
            const unsigned short* p = vb + (j * 16 + nc) * T_ + half * 8;
            bv.q[0] = *(const uint4*)(p);
            bv.q[1] = *(const uint4*)(p + 16);
            o[j] = wmma_bf16(pf, bv, o[j]);
        }
    }

    int bidx = bh >> 4, h = bh & 15;
#pragma unroll
    for (int r = 0; r < 8; ++r) {
        float inv = 1.0f / lr[r];
        int t = q0 + r + 8 * half;
        unsigned short* row = Ob + (size_t)(bidx * T_ + t) * D_ + h * HD_;
#pragma unroll
        for (int j = 0; j < 8; ++j) row[j * 16 + nc] = f2bf(o[j][r] * inv);
    }
}

// ---------------- output projection ----------------
// One wave computes 16x64 fp32 output. tasks = 256 m-tiles * 32 n-groups = 8192.
__global__ __launch_bounds__(128) void gemm_out(
    const unsigned short* __restrict__ abf,   // [B*T, D] bf16 (attention out)
    const unsigned short* __restrict__ wT,    // [D, D] bf16 (transposed)
    float* __restrict__ out)                  // [B*T, D] fp32
{
    int lane = threadIdx.x & 31, wid = threadIdx.x >> 5;
    int task = blockIdx.x * 4 + wid;
    int mt = task & 255;
    int ng = task >> 8;           // 0..31
    int m0 = mt * 16, n0 = ng * 64;

    v8f zero = {0.f,0.f,0.f,0.f,0.f,0.f,0.f,0.f};
    v8f acc[4];
#pragma unroll
    for (int j = 0; j < 4; ++j) acc[j] = zero;

    const unsigned short* ab = abf + m0 * D_;
    for (int kc = 0; kc < D_ / 32; ++kc) {
        if (kc + 1 < D_ / 32)
            __builtin_prefetch(ab + (lane & 15) * D_ + (kc + 1) * 32, 0, 1);
        frag16 a = load_a_frag(ab, D_, kc * 32, lane);
#pragma unroll
        for (int j = 0; j < 4; ++j) {
            frag16 b = load_b_frag(wT + (n0 + j * 16) * D_, D_, kc * 32, lane);
            acc[j] = wmma_bf16(a, b, acc[j]);
        }
    }
    int half = lane >> 4, nc = lane & 15;
#pragma unroll
    for (int r = 0; r < 8; ++r) {
        int m = m0 + r + 8 * half;
#pragma unroll
        for (int j = 0; j < 4; ++j) out[(size_t)m * D_ + n0 + j * 16 + nc] = acc[j][r];
    }
}

extern "C" void kernel_launch(void* const* d_in, const int* in_sizes, int n_in,
                              void* d_out, int out_size, void* d_ws, size_t ws_size,
                              hipStream_t stream) {
    const float* x     = (const float*)d_in[0];   // [B,T,D]
    const float* w_qkv = (const float*)d_in[1];   // [D,3D]
    const float* w_out = (const float*)d_in[2];   // [D,D]
    float* out = (float*)d_out;

    char* ws = (char*)d_ws;
    size_t off = 0;
    unsigned short* x_bf  = (unsigned short*)(ws + off); off += (size_t)B_ * T_ * D_ * 2;       // 16 MB
    unsigned short* wqkvT = (unsigned short*)(ws + off); off += (size_t)D_ * 3 * D_ * 2;        // 24 MB
    unsigned short* woutT = (unsigned short*)(ws + off); off += (size_t)D_ * D_ * 2;            //  8 MB
    unsigned short* Qb    = (unsigned short*)(ws + off); off += (size_t)B_ * H_ * T_ * HD_ * 2; // 16 MB
    unsigned short* Kb    = (unsigned short*)(ws + off); off += (size_t)B_ * H_ * T_ * HD_ * 2; // 16 MB
    unsigned short* Vt    = (unsigned short*)(ws + off); off += (size_t)B_ * H_ * T_ * HD_ * 2; // 16 MB
    unsigned short* Ob    = (unsigned short*)(ws + off); off += (size_t)B_ * T_ * D_ * 2;       // 16 MB

    cvt_bf16<<<2048, 256, 0, stream>>>(x, x_bf, B_ * T_ * D_);
    cvt_transpose<<<4096, 256, 0, stream>>>(w_qkv, wqkvT, D_, 3 * D_);
    cvt_transpose<<<2048, 256, 0, stream>>>(w_out, woutT, D_, D_);

    gemm_qkv_rope<<<3072, 128, 0, stream>>>(x_bf, wqkvT, Qb, Kb, Vt);   // 12288 wave-tasks
    attn_fwd<<<1024, 128, 0, stream>>>(Qb, Kb, Vt, Ob);                 // 4096 wave-tasks
    gemm_out<<<2048, 128, 0, stream>>>(Ob, woutT, out);                 // 8192 wave-tasks
}
